// GAN_Module_17463337026274
// MI455X (gfx1250) — matrix operand order repaired and checked by reading, not compiled, not run on hardware
//
#include <hip/hip_runtime.h>
#include <stdint.h>

#define B_      2
#define N_      8192
#define EN_     16
#define C_      256
#define F_      512
#define HEADS_  8
#define DE_     8
#define NPTS    (B_*N_)

typedef _Float16 h16;
typedef __attribute__((ext_vector_type(16))) _Float16 v16h;
typedef __attribute__((ext_vector_type(8)))  float    v8f;
typedef unsigned int u32x4 __attribute__((ext_vector_type(4)));
typedef int          i32x8 __attribute__((ext_vector_type(8)));
typedef int          i32x4 __attribute__((ext_vector_type(4)));

#if defined(__has_builtin)
#if __has_builtin(__builtin_amdgcn_tensor_load_to_lds) && __has_builtin(__builtin_amdgcn_s_wait_tensorcnt)
#define USE_TDM 1
#endif
#endif

// ---------------- WMMA fragment helpers (gfx1250 16x16x32 f16 layouts) ----------------
// A (16x32 f16): lane&15 = row M; lanes<16 hold K 0-7 & 16-23, lanes>=16 hold K 8-15 & 24-31.
__device__ __forceinline__ v16h frag_a_lds(const h16* A, int ldk, int k0, int lane) {
  const int m = lane & 15, hi = lane >> 4;
  const h16* p = A + m * ldk + k0 + hi * 8;
  v16h a;
#pragma unroll
  for (int i = 0; i < 8; ++i) { a[i] = p[i]; a[8 + i] = p[16 + i]; }
  return a;
}
// B (32x16 f16) from N-major (transposed) weights Wt[N][Kpad]:
// lane&15 = col N; lanes<16 hold K 0-7 & 16-23, lanes>=16 hold K 8-15 & 24-31.
__device__ __forceinline__ v16h frag_b_glb(const h16* Wt, int ldk, int n0, int k0, int lane) {
  const h16* p = Wt + (size_t)(n0 + (lane & 15)) * ldk + k0 + ((lane >> 4) << 3);
  v16h b;
#pragma unroll
  for (int i = 0; i < 8; ++i) { b[i] = p[i]; b[8 + i] = p[16 + i]; }
  return b;
}
__device__ __forceinline__ v8f wmma_f16(v16h a, v16h b, v8f c) {
  return __builtin_amdgcn_wmma_f32_16x16x32_f16(false, a, false, b, (short)0, c, false, false);
}
// D/C f32 16x16: lane l, vgpr r -> M = 8*(l>>4)+r, N = l&15.

// ---------------- weight prep: fp32 [K][N] -> f16 [Npad][Kpad] (zero padded) ----------------
__global__ void kw_prep(const float* __restrict__ W, h16* __restrict__ Wt,
                        int K, int Nfull, int Kpad, int Npad) {
  int idx = blockIdx.x * blockDim.x + threadIdx.x;
  if (idx >= Npad * Kpad) return;
  int n = idx / Kpad, k = idx % Kpad;
  float v = (k < K && n < Nfull) ? W[(size_t)k * Nfull + n] : 0.f;
  Wt[idx] = (h16)v;
}

// ---------------- K1: LayerNorm1, (B,C,N) -> f16 rows (B*N, C) ----------------
__global__ __launch_bounds__(256) void k1_ln1(const float* __restrict__ features,
                                              const float* __restrict__ g1,
                                              const float* __restrict__ be1,
                                              h16* __restrict__ x16) {
  int gp = blockIdx.x * 256 + threadIdx.x;
  int b = gp / N_, n = gp % N_;
  const float* col = features + (size_t)b * C_ * N_ + n;
  float sum = 0.f, ss = 0.f;
  for (int c = 0; c < C_; ++c) { float v = col[(size_t)c * N_]; sum += v; ss += v * v; }
  float mean = sum * (1.f / 256.f);
  float var  = ss * (1.f / 256.f) - mean * mean;
  float rs   = rsqrtf(var + 1e-6f);
  h16* dst = x16 + (size_t)gp * C_;
  for (int c = 0; c < C_; ++c) {
    float v = col[(size_t)c * N_];
    dst[c] = (h16)((v - mean) * rs * g1[c] + be1[c]);
  }
}

// ---------------- K2: per-point pipeline (2 points / workgroup, 4 waves) ----------------
struct __align__(16) SMemK2 {
  h16   sf[2][16][256];    // gathered x rows (TDM gather target)
  h16   xown[2][256];      // each point's own x row
  h16   hmid[2][16][128];  // relu(sf@D1+db1)
  h16   sA[16][288];       // combined s matrix (2 pts x 8 slots), K padded 259->288
  float hval[2][16][8];    // edge-MLP logits h[e][d]
  float wgt[2][8][16];     // gumbel-softmax weights
  float scg[2][16][3];     // gathered xyz
  float kmat[16][256];
  float vmat[16][256];
  float qv[2][256];
  float simi[2][8][8];
  float sme[2][8][8];
  float ov[2][256];
};

__global__ __launch_bounds__(128) void k2_point(
    const h16* __restrict__ x16, const float* __restrict__ xyz,
    const float* __restrict__ features, const int* __restrict__ edges,
    const float* __restrict__ gum,
    const h16* __restrict__ D1t, const float* __restrict__ db1,
    const h16* __restrict__ D2t, const float* __restrict__ db2,
    const h16* __restrict__ Wqt, const h16* __restrict__ Wkt,
    const h16* __restrict__ Wvt, const h16* __restrict__ Wot,
    float* __restrict__ feats1) {
  __shared__ SMemK2 sm;
  const int t = threadIdx.x, wave = t >> 5, lane = t & 31;
  const int p0 = blockIdx.x * 2;

  // ---- Phase 0: gathers ----
#ifdef USE_TDM
  // TDM gather mode: 16 row indices per descriptor == EN exactly. One TDM op per point
  // (waves 0 and 1), rows of 512B (data_size=8B units, 64 units/row) into sf[p].
  if (wave < 2) {
    int gp = p0 + wave, b = gp / N_, n = gp % N_;
    const int* ebase = edges + ((size_t)b * N_ * EN_ + (size_t)n * EN_) * 2;
    unsigned int pk[8];
#pragma unroll
    for (int i = 0; i < 8; ++i) {
      unsigned int i0 = (unsigned int)ebase[(2 * i + 0) * 2];
      unsigned int i1 = (unsigned int)ebase[(2 * i + 1) * 2];
      pk[i] = (i0 & 0xFFFFu) | (i1 << 16);
    }
    uint64_t gaddr = (uint64_t)(uintptr_t)(x16 + (size_t)b * N_ * C_);
    unsigned int ldsoff =
        (unsigned int)(__builtin_offsetof(SMemK2, sf) + wave * (16 * 256 * 2));
    u32x4 g0; i32x8 g1; i32x4 g2, g3; i32x8 gx = {};
    g0[0] = 0x80000001u;                                   // count=1, gather_mode, 16-bit idx
    g0[1] = ldsoff;                                        // lds_addr
    g0[2] = (unsigned int)gaddr;                           // global_addr[31:0]
    g0[3] = ((unsigned int)(gaddr >> 32) & 0x01FFFFFFu) | (2u << 30); // addr[56:32]|type=2
    g1[0] = (3 << 16);                                     // data_size = 8 bytes
    g1[1] = (int)(64u << 16);                              // tensor_dim0 = 64 units (512B row)
    g1[2] = (int)((unsigned)N_ << 16);                     // tensor_dim1 = 8192 rows
    g1[3] = (int)(64u << 16);                              // tile_dim0 = 64 (full row)
    g1[4] = 16;                                            // tile_dim1 = 16 valid indices
    g1[5] = 64;                                            // tensor_dim0_stride = 64 units
    g1[6] = 0; g1[7] = 0;
    g2[0] = (int)pk[0]; g2[1] = (int)pk[1]; g2[2] = (int)pk[2]; g2[3] = (int)pk[3];
    g3[0] = (int)pk[4]; g3[1] = (int)pk[5]; g3[2] = (int)pk[6]; g3[3] = (int)pk[7];
    __builtin_amdgcn_tensor_load_to_lds(g0, g1, g2, g3, gx, 0);
    __builtin_amdgcn_s_wait_tensorcnt(0);
  }
#else
  for (int r = wave; r < 32; r += 4) {   // fallback: vectorized per-wave row gather
    int p = r >> 4, e = r & 15;
    int gp = p0 + p, b = gp / N_, n = gp % N_;
    int idx = edges[((size_t)b * N_ * EN_ + (size_t)n * EN_ + e) * 2];
    const uint4* src = (const uint4*)(x16 + ((size_t)b * N_ + idx) * C_);
    ((uint4*)&sm.sf[p][e][0])[lane] = src[lane];
  }
#endif
  { // own x row (for q)
    int p = t >> 6, s = t & 63;
    ((uint2*)&sm.xown[p][0])[s] = ((const uint2*)(x16 + (size_t)(p0 + p) * C_))[s];
  }
  if (t < 32) { // gathered coords
    int p = t >> 4, e = t & 15;
    int gp = p0 + p, b = gp / N_, n = gp % N_;
    int idx = edges[((size_t)b * N_ * EN_ + (size_t)n * EN_ + e) * 2];
    const float* src = xyz + ((size_t)b * N_ + idx) * 3;
    sm.scg[p][e][0] = src[0]; sm.scg[p][e][1] = src[1]; sm.scg[p][e][2] = src[2];
  }
  __syncthreads();

  // ---- Edge MLP GEMM1: (16x256)@D1t(128x256) per point, relu ----
  {
    int p = wave >> 1, nt0 = (wave & 1) * 4;
    for (int nt = nt0; nt < nt0 + 4; ++nt) {
      v8f acc = {};
#pragma unroll
      for (int kt = 0; kt < 8; ++kt) {
        v16h a  = frag_a_lds(&sm.sf[p][0][0], 256, kt * 32, lane);
        v16h bb = frag_b_glb(D1t, 256, nt * 16, kt * 32, lane);
        acc = wmma_f16(a, bb, acc);
      }
      int cc = nt * 16 + (lane & 15);
      float bia = db1[cc];
#pragma unroll
      for (int r = 0; r < 8; ++r) {
        int m = ((lane >> 4) << 3) + r;
        float v = acc[r] + bia; v = v > 0.f ? v : 0.f;
        sm.hmid[p][m][cc] = (h16)v;
      }
    }
  }
  __syncthreads();

  // ---- Edge MLP GEMM2: (16x128)@D2t(16x128, N padded 8->16) ----
  if (wave < 2) {
    int p = wave;
    v8f acc = {};
#pragma unroll
    for (int kt = 0; kt < 4; ++kt) {
      v16h a  = frag_a_lds(&sm.hmid[p][0][0], 128, kt * 32, lane);
      v16h bb = frag_b_glb(D2t, 128, 0, kt * 32, lane);
      acc = wmma_f16(a, bb, acc);
    }
    int cc = lane & 15;
    if (cc < 8) {
      float bia = db2[cc];
#pragma unroll
      for (int r = 0; r < 8; ++r)
        sm.hval[p][((lane >> 4) << 3) + r][cc] = acc[r] + bia;
    }
  }
  __syncthreads();

  // ---- Gumbel softmax over EN=16 (tau=0.1) ----
  if (t < 16) {
    int p = t >> 3, d = t & 7, gp = p0 + p;
    const float* u = gum + ((size_t)gp * DE_ + d) * EN_;
    float z[16]; float mx = -1e30f;
    for (int e = 0; e < 16; ++e) {
      float gn = -logf(-logf(u[e]));
      float val = (sm.hval[p][e][d] + gn) * 10.0f;
      z[e] = val; mx = val > mx ? val : mx;
    }
    float ssum = 0.f;
    for (int e = 0; e < 16; ++e) { float ex = expf(z[e] - mx); z[e] = ex; ssum += ex; }
    float inv = 1.f / ssum;
    for (int e = 0; e < 16; ++e) sm.wgt[p][d][e] = z[e] * inv;
  }
  __syncthreads();

  // ---- Aggregation: build s = [w@sf | w@sc - xyz | 0pad] as combined 16x288 f16 ----
  {
    int r = t >> 3, s8 = t & 7;
    int p = r >> 3, d = r & 7;
    int gp = p0 + p, b = gp / N_, n = gp % N_;
    float wrow[16];
#pragma unroll
    for (int e = 0; e < 16; ++e) wrow[e] = sm.wgt[p][d][e];
    for (int c = s8; c < 256; c += 8) {
      float acc = 0.f;
#pragma unroll
      for (int e = 0; e < 16; ++e) acc += wrow[e] * (float)sm.sf[p][e][c];
      sm.sA[r][c] = (h16)acc;
    }
    if (s8 < 3) {
      float acc = 0.f;
#pragma unroll
      for (int e = 0; e < 16; ++e) acc += wrow[e] * sm.scg[p][e][s8];
      acc -= xyz[((size_t)b * N_ + n) * 3 + s8];
      sm.sA[r][256 + s8] = (h16)acc;
    }
    for (int c = 259 + s8; c < 288; c += 8) sm.sA[r][c] = (h16)0.f;
  }
  __syncthreads();

  // ---- K/V projections: (16x288)@Wk/Wv(256x288) via WMMA ----
  for (int which = 0; which < 2; ++which) {
    const h16* Wt = which ? Wvt : Wkt;
    float (*outm)[256] = which ? sm.vmat : sm.kmat;
    for (int nt = wave * 4; nt < wave * 4 + 4; ++nt) {
      v8f acc = {};
#pragma unroll
      for (int kt = 0; kt < 9; ++kt) {
        v16h a  = frag_a_lds(&sm.sA[0][0], 288, kt * 32, lane);
        v16h bb = frag_b_glb(Wt, 288, nt * 16, kt * 32, lane);
        acc = wmma_f16(a, bb, acc);
      }
      int cc = nt * 16 + (lane & 15);
#pragma unroll
      for (int r = 0; r < 8; ++r) outm[((lane >> 4) << 3) + r][cc] = acc[r];
    }
  }
  // ---- q = x @ Wq (M=1 per point; VALU) ----
  for (int o = t; o < 512; o += 128) {
    int p = o >> 8, c = o & 255;
    float acc = 0.f;
    const h16* wq = Wqt + (size_t)c * 256;
    for (int k = 0; k < 256; ++k) acc += (float)sm.xown[p][k] * (float)wq[k];
    sm.qv[p][c] = acc;
  }
  __syncthreads();

  // ---- attention over DE=8 slots, 8 heads x 32 ----
  {
    int p = t >> 6, de = (t >> 3) & 7, h = t & 7;
    float acc = 0.f;
#pragma unroll
    for (int d2 = 0; d2 < 32; ++d2)
      acc += sm.kmat[p * 8 + de][h * 32 + d2] * sm.qv[p][h * 32 + d2];
    sm.simi[p][de][h] = acc * 0.17677669529663687f;   // 1/sqrt(32)
  }
  __syncthreads();
  if (t < 16) {
    int p = t >> 3, h = t & 7;
    float mx = -1e30f;
    for (int de = 0; de < 8; ++de) mx = fmaxf(mx, sm.simi[p][de][h]);
    float ssum = 0.f;
    for (int de = 0; de < 8; ++de) {
      float e2 = expf(sm.simi[p][de][h] - mx);
      sm.sme[p][de][h] = e2; ssum += e2;
    }
    float inv = 1.f / ssum;
    for (int de = 0; de < 8; ++de) sm.sme[p][de][h] *= inv;
  }
  __syncthreads();
  for (int o = t; o < 512; o += 128) {
    int p = o >> 8, c = o & 255, h = c >> 5;
    float acc = 0.f;
#pragma unroll
    for (int de = 0; de < 8; ++de) acc += sm.vmat[p * 8 + de][c] * sm.sme[p][de][h];
    sm.ov[p][c] = acc;
  }
  __syncthreads();

  // ---- out = o@Wo + residual(features) -> feats1 ----
  for (int o = t; o < 512; o += 128) {
    int p = o >> 8, c = o & 255;
    int gp = p0 + p, b = gp / N_, n = gp % N_;
    float acc = 0.f;
    const h16* wo = Wot + (size_t)c * 256;
    for (int k = 0; k < 256; ++k) acc += sm.ov[p][k] * (float)wo[k];
    feats1[(size_t)gp * C_ + c] = features[((size_t)b * C_ + c) * N_ + n] + acc;
  }
}

// ---------------- K3: LN2 + FFN (WMMA) + residual + transposed store ----------------
struct __align__(16) SMemK3 {
  h16   xf[16][256];
  h16   a1[16][512];
  float fres[16][256];
};

__global__ __launch_bounds__(128) void k3_ffn(
    const float* __restrict__ feats1,
    const float* __restrict__ g2v, const float* __restrict__ be2,
    const h16* __restrict__ W1t, const float* __restrict__ bf1,
    const h16* __restrict__ W2t, const float* __restrict__ bf2,
    float* __restrict__ out) {
  __shared__ SMemK3 sm;
  const int t = threadIdx.x, wave = t >> 5, lane = t & 31;
  // LN2: 8 threads per row, shuffle reduce within 8-lane groups
  {
    int r = t >> 3, s = t & 7;
    int gp = blockIdx.x * 16 + r;
    const float* src = feats1 + (size_t)gp * C_;
    float vals[32]; float sum = 0.f, ss = 0.f;
#pragma unroll
    for (int j = 0; j < 32; ++j) { float v = src[s + 8 * j]; vals[j] = v; sum += v; ss += v * v; }
    for (int m = 1; m < 8; m <<= 1) { sum += __shfl_xor(sum, m, 8); ss += __shfl_xor(ss, m, 8); }
    float mean = sum * (1.f / 256.f);
    float var  = ss * (1.f / 256.f) - mean * mean;
    float rs   = rsqrtf(var + 1e-6f);
#pragma unroll
    for (int j = 0; j < 32; ++j) {
      int c = s + 8 * j;
      sm.fres[r][c] = vals[j];
      sm.xf[r][c] = (h16)((vals[j] - mean) * rs * g2v[c] + be2[c]);
    }
  }
  __syncthreads();
  // GEMM1: (16x256)@W1t(512x256), relu
  for (int nt = wave * 8; nt < wave * 8 + 8; ++nt) {
    v8f acc = {};
#pragma unroll
    for (int kt = 0; kt < 8; ++kt) {
      v16h a  = frag_a_lds(&sm.xf[0][0], 256, kt * 32, lane);
      v16h bb = frag_b_glb(W1t, 256, nt * 16, kt * 32, lane);
      acc = wmma_f16(a, bb, acc);
    }
    int cc = nt * 16 + (lane & 15);
    float bia = bf1[cc];
#pragma unroll
    for (int rr = 0; rr < 8; ++rr) {
      float v = acc[rr] + bia; v = v > 0.f ? v : 0.f;
      sm.a1[((lane >> 4) << 3) + rr][cc] = (h16)v;
    }
  }
  __syncthreads();
  // GEMM2: (16x512)@W2t(256x512) + bias + residual, write (B,C,N)
  for (int nt = wave * 4; nt < wave * 4 + 4; ++nt) {
    v8f acc = {};
#pragma unroll
    for (int kt = 0; kt < 16; ++kt) {
      v16h a  = frag_a_lds(&sm.a1[0][0], 512, kt * 32, lane);
      v16h bb = frag_b_glb(W2t, 512, nt * 16, kt * 32, lane);
      acc = wmma_f16(a, bb, acc);
    }
    int cc = nt * 16 + (lane & 15);
    float bia = bf2[cc];
#pragma unroll
    for (int rr = 0; rr < 8; ++rr) {
      int m = ((lane >> 4) << 3) + rr;
      int gp2 = blockIdx.x * 16 + m;
      int b = gp2 / N_, n = gp2 % N_;
      out[((size_t)b * C_ + cc) * N_ + n] = acc[rr] + bia + sm.fres[m][cc];
    }
  }
}

// ---------------- launch ----------------
extern "C" void kernel_launch(void* const* d_in, const int* in_sizes, int n_in,
                              void* d_out, int out_size, void* d_ws, size_t ws_size,
                              hipStream_t stream) {
  (void)in_sizes; (void)n_in; (void)out_size; (void)ws_size;
  const float* xyz      = (const float*)d_in[0];
  const float* features = (const float*)d_in[1];
  const int*   edges    = (const int*)d_in[2];
  const float* gum      = (const float*)d_in[3];
  const float* g1       = (const float*)d_in[4];
  const float* be1      = (const float*)d_in[5];
  const float* Wq       = (const float*)d_in[6];
  const float* Wk       = (const float*)d_in[7];
  const float* Wv       = (const float*)d_in[8];
  const float* Wo       = (const float*)d_in[9];
  const float* g2       = (const float*)d_in[10];
  const float* be2      = (const float*)d_in[11];
  const float* W1       = (const float*)d_in[12];
  const float* bf1      = (const float*)d_in[13];
  const float* W2       = (const float*)d_in[14];
  const float* bf2      = (const float*)d_in[15];
  const float* D1       = (const float*)d_in[16];
  const float* db1      = (const float*)d_in[17];
  const float* D2       = (const float*)d_in[18];
  const float* db2      = (const float*)d_in[19];
  float* out = (float*)d_out;

  uint8_t* ws = (uint8_t*)d_ws;
  size_t off = 0;
  auto alloc = [&](size_t bytes) -> void* {
    void* p = ws + off;
    off += (bytes + 255) & ~(size_t)255;
    return p;
  };
  h16*   x16    = (h16*)alloc((size_t)NPTS * C_ * 2);
  float* feats1 = (float*)alloc((size_t)NPTS * C_ * 4);
  h16* Wqt = (h16*)alloc(256 * 256 * 2);
  h16* Wkt = (h16*)alloc(256 * 288 * 2);
  h16* Wvt = (h16*)alloc(256 * 288 * 2);
  h16* Wot = (h16*)alloc(256 * 256 * 2);
  h16* W1t = (h16*)alloc(512 * 256 * 2);
  h16* W2t = (h16*)alloc(256 * 512 * 2);
  h16* D1t = (h16*)alloc(128 * 256 * 2);
  h16* D2t = (h16*)alloc(16 * 128 * 2);

  auto prep = [&](const float* W, h16* Wt, int K, int Nf, int Kp, int Np) {
    int tot = Np * Kp;
    kw_prep<<<(tot + 255) / 256, 256, 0, stream>>>(W, Wt, K, Nf, Kp, Np);
  };
  prep(Wq, Wqt, 256, 256, 256, 256);
  prep(Wk, Wkt, 259, 256, 288, 256);
  prep(Wv, Wvt, 259, 256, 288, 256);
  prep(Wo, Wot, 256, 256, 256, 256);
  prep(W1, W1t, 256, 512, 256, 512);
  prep(W2, W2t, 512, 256, 512, 256);
  prep(D1, D1t, 256, 128, 256, 128);
  prep(D2, D2t, 128, 8, 128, 16);

  k1_ln1<<<NPTS / 256, 256, 0, stream>>>(features, g1, be1, x16);
  k2_point<<<NPTS / 2, 128, 0, stream>>>(x16, xyz, features, edges, gum,
                                         D1t, db1, D2t, db2,
                                         Wqt, Wkt, Wvt, Wot, feats1);
  k3_ffn<<<NPTS / 16, 128, 0, stream>>>(feats1, g2, be2, W1t, bf1, W2t, bf2, out);
}
